// SRC_classifier_66778151518219
// MI455X (gfx1250) — matrix operand order repaired and checked by reading, not compiled, not run on hardware
//
#include <hip/hip_runtime.h>
#include <math.h>

#define N_NODES 10000
#define N_EDGES 160000

typedef __attribute__((ext_vector_type(16))) __bf16 v16bf;
typedef __attribute__((ext_vector_type(8)))  __bf16 v8bf;
typedef __attribute__((ext_vector_type(8)))  float  v8f;

// ---------- monotone float<->uint encoding for atomicMax-based segment max ----
__device__ __forceinline__ unsigned fenc(float f) {
  unsigned u = __float_as_uint(f);
  return (u & 0x80000000u) ? ~u : (u | 0x80000000u);
}
__device__ __forceinline__ float fdec(unsigned u) {
  return __uint_as_float((u & 0x80000000u) ? (u & 0x7FFFFFFFu) : ~u);
}

// ---------- A fragment loaders -----------------------------------------------
// ISA 16-bit A layout: lanes 0-15 = rows, hold K {0..7, 16..23};
// lanes 16-31 = rows, hold K {8..15, 24..31}.  Element e=2r+h:
//   r<4 -> K = koff + 2r + h,  r>=4 -> K = koff + 16 + 2(r-4) + h.
__device__ __forceinline__ v16bf load_a_fast(const float* __restrict__ ap) {
  float2 f[8];
#pragma unroll
  for (int j = 0; j < 4; ++j) f[j]     = *(const float2*)(ap + 2 * j);
#pragma unroll
  for (int j = 0; j < 4; ++j) f[4 + j] = *(const float2*)(ap + 16 + 2 * j);
  v16bf a;
#pragma unroll
  for (int j = 0; j < 8; ++j) {
    a[2 * j]     = (__bf16)f[j].x;
    a[2 * j + 1] = (__bf16)f[j].y;
  }
  return a;
}

// Checked path: only used for the final (partial) K-step.
__device__ __forceinline__ v16bf load_a_checked(const float* __restrict__ arow,
                                                int kBase, int K) {
  const int lane = threadIdx.x & 31;
  const int koff = (lane & 16) ? 8 : 0;
  v16bf a;
#pragma unroll
  for (int r = 0; r < 8; ++r) {
    const int kb = (r < 4) ? (2 * r) : (16 + 2 * (r - 4));
#pragma unroll
    for (int h = 0; h < 2; ++h) {
      const int k = kBase + kb + koff + h;
      const float f = (k < K) ? arow[k] : 0.0f;
      a[2 * r + h] = (__bf16)f;
    }
  }
  return a;
}

// ---------- fused Q|K|V|S projection GEMM: out[M, 4*dout] = A[M,K] @ W4 + b ---
// Block = 256 threads (8 waves). Each wave computes a 16-row x 256-col tile:
// 16 accumulators, 16 WMMAs per K-step reusing one A fragment (A HBM traffic
// cut 4x vs 64-wide strips; weights live in L2). B tile (32k x 256n) staged in
// LDS as bf16 (one column per thread, 4 packed b128 stores). All bounds logic
// hoisted out of the K loop.
#define BSTRIDE 40   // padded k-stride (bf16 units): 80B rows, 16B-aligned frags
#define NTILES  16   // 16 n-tiles of 16 = 256 columns per block
__global__ void __launch_bounds__(256) k_gemm_qkvs(
    const float* __restrict__ A, int M, int K,
    const float* __restrict__ Wq, const float* __restrict__ Wk,
    const float* __restrict__ Wv, const float* __restrict__ Ws,
    const float* __restrict__ bq, const float* __restrict__ bk,
    const float* __restrict__ bv, const float* __restrict__ bs,
    int dout, float* __restrict__ out) {
  __shared__ __bf16 Bs[256][BSTRIDE];  // [n-local][k-local]

  const int N4   = 4 * dout;
  const int wave = threadIdx.x >> 5;
  const int lane = threadIdx.x & 31;

  const int mTiles   = M >> 4;                   // M is a multiple of 16
  const int mTileRaw = blockIdx.x * 8 + wave;
  const bool active  = (mTileRaw < mTiles);
  const int mTile    = active ? mTileRaw : (mTiles - 1);  // clamp: keep waves alive
  const int mBase    = mTile * 16;
  const int nBase0   = blockIdx.y * 256;

  // ---- loop-invariant B-staging setup: this thread owns one column (nCol) and
  //      stages all 32 k-values per step. Invalid columns read a dummy column
  //      (Wq col 0) and multiply by 0 -> branch-free hot loop.
  const int col256 = threadIdx.x;                // 0..255
  const int nCol   = nBase0 + col256;
  const bool nValid = (nCol < N4);
  const int whichC  = nValid ? (nCol / dout) : 0;
  const int cC      = nValid ? (nCol - whichC * dout) : 0;
  const float* Wsel = (whichC == 0) ? Wq : (whichC == 1) ? Wk
                    : (whichC == 2) ? Wv : Ws;
  const float* colPtr = Wsel + cC;
  const float nMul = nValid ? 1.0f : 0.0f;

  const float* arow = A + (size_t)(mBase + (lane & 15)) * K;
  const int koff = (lane & 16) ? 8 : 0;
  const int kFragOff = (lane & 16) ? 8 : 0;
  const int colLane  = lane & 15;

  v8f acc[NTILES] = {};
  const int Kmain = K & ~31;

  for (int kBase = 0; kBase < Kmain; kBase += 32) {
    // ---- stage B: 32 unguarded strided loads, 4 packed b128 LDS stores ------
#pragma unroll
    for (int q8 = 0; q8 < 4; ++q8) {
      v8bf pack;
#pragma unroll
      for (int j = 0; j < 8; ++j) {
        const float v = colPtr[(size_t)(kBase + q8 * 8 + j) * dout] * nMul;
        pack[j] = (__bf16)v;
      }
      *(v8bf*)&Bs[col256][q8 * 8] = pack;
    }
    __syncthreads();

    // ---- A fragment: wide branch-free loads, reused by 16 WMMAs -------------
    const v16bf a = load_a_fast(arow + kBase + koff);
    __builtin_prefetch(arow + kBase + 32 + koff, 0, 1);  // global_prefetch_b8

#pragma unroll
    for (int t = 0; t < NTILES; ++t) {
      const __bf16* bp = &Bs[16 * t + colLane][kFragOff];  // 2x8 bf16 at k 8h and 16+8h, 16B-aligned
      v16bf b;
#pragma unroll
      for (int e = 0; e < 16; ++e) b[e] = bp[(e < 8) ? e : (8 + e)];
      acc[t] = __builtin_amdgcn_wmma_f32_16x16x32_bf16(
          false, a, false, b, (short)0, acc[t], false, false);
    }
    __syncthreads();
  }

  if (Kmain < K) {  // one guarded tail step (uniform condition)
#pragma unroll
    for (int q8 = 0; q8 < 4; ++q8) {
      v8bf pack;
#pragma unroll
      for (int j = 0; j < 8; ++j) {
        const int k = Kmain + q8 * 8 + j;
        const float v = (k < K) ? colPtr[(size_t)k * dout] * nMul : 0.0f;
        pack[j] = (__bf16)v;
      }
      *(v8bf*)&Bs[col256][q8 * 8] = pack;
    }
    __syncthreads();

    const v16bf a = load_a_checked(arow, Kmain, K);
#pragma unroll
    for (int t = 0; t < NTILES; ++t) {
      const __bf16* bp = &Bs[16 * t + colLane][kFragOff];
      v16bf b;
#pragma unroll
      for (int e = 0; e < 16; ++e) b[e] = bp[(e < 8) ? e : (8 + e)];
      acc[t] = __builtin_amdgcn_wmma_f32_16x16x32_bf16(
          false, a, false, b, (short)0, acc[t], false, false);
    }
  }

  if (!active) return;

  // Epilogue: C/D layout — lane<16: M=r, N=lane; lane>=16: M=r+8, N=lane-16.
  const int rowOff = (lane & 16) ? 8 : 0;
#pragma unroll
  for (int t = 0; t < NTILES; ++t) {
    const int n = nBase0 + 16 * t + colLane;
    if (n >= N4) continue;
    const int which = n / dout;
    const int col   = n - which * dout;
    const float* bias = (which == 0) ? bq : (which == 1) ? bk
                      : (which == 2) ? bv : bs;
    const float bval = bias[col];
#pragma unroll
    for (int r = 0; r < 8; ++r) {
      const int row = mBase + r + rowOff;
      out[(size_t)row * N4 + n] = acc[t][r] + bval;
    }
  }
}

// ---------- edge / softmax / scatter pipeline -------------------------------
__global__ void k_init_nodes(unsigned* __restrict__ mEnc, float* __restrict__ den, int n) {
  const int i = blockIdx.x * blockDim.x + threadIdx.x;
  if (i < n) { mEnc[i] = fenc(-INFINITY); den[i] = 0.0f; }
}

__global__ void k_zero(float* __restrict__ p, long n) {
  const long i = (long)blockIdx.x * blockDim.x + threadIdx.x;
  if (i < n) p[i] = 0.0f;
}

__global__ void k_logits(const float* __restrict__ QKVS,
                         const int* __restrict__ src, const int* __restrict__ dst,
                         int dout, float scale,
                         float* __restrict__ logits, unsigned* __restrict__ mEnc) {
  const int e = blockIdx.x * blockDim.x + threadIdx.x;
  if (e >= N_EDGES) return;
  const int s = src[e], d = dst[e];
  const int N4 = 4 * dout;
  const float* q = QKVS + (size_t)d * N4;            // Q row of dst
  const float* k = QKVS + (size_t)s * N4 + dout;     // K row of src
  float acc = 0.0f;
  for (int f = 0; f < dout; ++f) acc += q[f] * k[f];
  acc *= scale;
  logits[e] = acc;
  atomicMax(&mEnc[d], fenc(acc));
}

__global__ void k_expsum(const float* __restrict__ logits, const int* __restrict__ dst,
                         const unsigned* __restrict__ mEnc,
                         float* __restrict__ expe, float* __restrict__ den) {
  const int e = blockIdx.x * blockDim.x + threadIdx.x;
  if (e >= N_EDGES) return;
  const int d = dst[e];
  const float ex = expf(logits[e] - fdec(mEnc[d]));
  expe[e] = ex;
  atomicAdd(&den[d], ex);
}

__global__ void k_agg(const float* __restrict__ QKVS,
                      const int* __restrict__ src, const int* __restrict__ dst, int dout,
                      const float* __restrict__ expe, const float* __restrict__ den,
                      float* __restrict__ agg) {
  const long idx = (long)blockIdx.x * blockDim.x + threadIdx.x;
  const long total = (long)N_EDGES * dout;
  if (idx >= total) return;
  const int e = (int)(idx / dout);
  const int f = (int)(idx - (long)e * dout);
  const int s = src[e], d = dst[e];
  const float alpha = expe[e] / fmaxf(den[d], 1e-16f);
  const float v = QKVS[(size_t)s * (4 * dout) + 2 * dout + f];
  atomicAdd(&agg[(size_t)d * dout + f], alpha * v);
}

// out = act(agg + skip); act: 0 = ELU, 1 = ReLU
__global__ void k_final(const float* __restrict__ QKVS, const float* __restrict__ agg,
                        int dout, int act, float* __restrict__ out) {
  const long idx = (long)blockIdx.x * blockDim.x + threadIdx.x;
  const long total = (long)N_NODES * dout;
  if (idx >= total) return;
  const int i = (int)(idx / dout);
  const int f = (int)(idx - (long)i * dout);
  float v = agg[idx] + QKVS[(size_t)i * (4 * dout) + 3 * dout + f];
  v = act ? fmaxf(v, 0.0f) : (v > 0.0f ? v : expm1f(v));
  out[idx] = v;
}

// ---------------------------------------------------------------------------
extern "C" void kernel_launch(void* const* d_in, const int* in_sizes, int n_in,
                              void* d_out, int out_size, void* d_ws, size_t ws_size,
                              hipStream_t stream) {
  (void)in_sizes; (void)n_in; (void)out_size; (void)ws_size;
  const float* x  = (const float*)d_in[0];
  const int*   ei = (const int*)d_in[1];
  const int* src = ei;            // edge_index[0]
  const int* dst = ei + N_EDGES;  // edge_index[1]

  // Workspace layout (floats)
  float* ws   = (float*)d_ws;
  float* H0   = ws;                                  // 10000*512
  float* H1   = H0 + (size_t)N_NODES * 512;          // 10000*512
  float* QKVS = H1 + (size_t)N_NODES * 512;          // 10000*2048
  float* AGG  = QKVS + (size_t)N_NODES * 2048;       // 10000*512
  float* LOG  = AGG + (size_t)N_NODES * 512;         // 160000
  float* EXPE = LOG + N_EDGES;                       // 160000
  unsigned* MENC = (unsigned*)(EXPE + N_EDGES);      // 10000
  float* DEN  = (float*)(MENC + N_NODES);            // 10000

  const int dims[5][2] = {{15962, 512}, {512, 64}, {64, 32}, {32, 16}, {16, 2}};
  float* Houts[2] = {H0, H1};

  const float* Acur = x;
  for (int l = 0; l < 5; ++l) {
    const int din = dims[l][0], dout = dims[l][1];
    const float* Wq = (const float*)d_in[2 + 8 * l + 0];
    const float* bq = (const float*)d_in[2 + 8 * l + 1];
    const float* Wk = (const float*)d_in[2 + 8 * l + 2];
    const float* bk = (const float*)d_in[2 + 8 * l + 3];
    const float* Wv = (const float*)d_in[2 + 8 * l + 4];
    const float* bv = (const float*)d_in[2 + 8 * l + 5];
    const float* Ws = (const float*)d_in[2 + 8 * l + 6];
    const float* bs = (const float*)d_in[2 + 8 * l + 7];

    // 1) fused projection GEMM (WMMA bf16, f32 accumulate)
    const int mTiles = N_NODES / 16;                 // 625
    const int N4 = 4 * dout;
    dim3 ggrid((mTiles + 7) / 8, (N4 + 255) / 256);
    k_gemm_qkvs<<<ggrid, 256, 0, stream>>>(Acur, N_NODES, din,
                                           Wq, Wk, Wv, Ws, bq, bk, bv, bs,
                                           dout, QKVS);

    // 2) segment softmax + aggregation
    k_init_nodes<<<(N_NODES + 255) / 256, 256, 0, stream>>>(MENC, DEN, N_NODES);
    const long aggN = (long)N_NODES * dout;
    k_zero<<<(unsigned)((aggN + 255) / 256), 256, 0, stream>>>(AGG, aggN);

    const float scale = 1.0f / sqrtf((float)dout);
    k_logits<<<(N_EDGES + 255) / 256, 256, 0, stream>>>(QKVS, src, dst, dout, scale, LOG, MENC);
    k_expsum<<<(N_EDGES + 255) / 256, 256, 0, stream>>>(LOG, dst, MENC, EXPE, DEN);

    const long en = (long)N_EDGES * dout;
    k_agg<<<(unsigned)((en + 255) / 256), 256, 0, stream>>>(QKVS, src, dst, dout, EXPE, DEN, AGG);

    // 3) skip + activation
    float* outp = (l == 4) ? (float*)d_out : Houts[l & 1];
    const int act = (l >= 2) ? 1 : 0;  // ELU for extractor layers, ReLU after
    k_final<<<(unsigned)((aggN + 255) / 256), 256, 0, stream>>>(QKVS, AGG, dout, act, outp);

    Acur = outp;
  }
}
